// MambaBackbone_48756468744432
// MI455X (gfx1250) — matrix operand order repaired and checked by reading, not compiled, not run on hardware
//
#include <hip/hip_runtime.h>
#include <hip/hip_bf16.h>
#include <stdint.h>

#define B_   16
#define T_   4096
#define DIN  256
#define H_   128
#define G3   384    // 3*H
#define LN_EPS 1e-5f

typedef __attribute__((ext_vector_type(16))) _Float16 v16h;
typedef __attribute__((ext_vector_type(8)))  float    v8f;

// ---------------------------------------------------------------------------
// One-shot f32 -> f16 weight conversion into workspace.
// ---------------------------------------------------------------------------
__global__ __launch_bounds__(256) void cvt_weights_kernel(const float* __restrict__ w_ih,
                                                          const float* __restrict__ w_hh,
                                                          const float* __restrict__ w_proj,
                                                          _Float16* __restrict__ wih16,
                                                          _Float16* __restrict__ whh16,
                                                          _Float16* __restrict__ wp16) {
    int i = blockIdx.x * 256 + threadIdx.x;
    if (i < G3 * DIN) wih16[i] = (_Float16)w_ih[i];
    if (i < G3 * H_)  whh16[i] = (_Float16)w_hh[i];
    if (i < H_ * H_)  wp16[i]  = (_Float16)w_proj[i];
}

// B-fragment: W16 row-major [N][K]; lane (n = lane&15) needs K = kk..kk+7 and
// 16+kk..16+kk+7 with kk = 8*(lane>>4) inside the 32-wide chunk -> two aligned 16B loads.
__device__ inline v16h load_b16(const _Float16* __restrict__ w16, int ldk,
                                int col0, int kc, int lane) {
    int n  = lane & 15;
    int kk = (lane >> 4) * 8;
    typedef __attribute__((ext_vector_type(8))) _Float16 v8h_;
    const _Float16* p = w16 + (size_t)(col0 + n) * ldk + kc + kk;
    v8h_ lo = *(const v8h_*)p;
    v8h_ hi = *(const v8h_*)(p + 16);
    return __builtin_shufflevector(lo, hi, 0,1,2,3,4,5,6,7,8,9,10,11,12,13,14,15);
}

// ---------------------------------------------------------------------------
// Phase 1: gi = x @ w_ih^T + b_ih   [65536 x 384], K=256.
// One 24-wave block per 16-row tile; x staged once in LDS as f16.
// ---------------------------------------------------------------------------
__global__ __launch_bounds__(768) void gi_kernel(const float* __restrict__ x,
                                                 const _Float16* __restrict__ wih16,
                                                 const float* __restrict__ b_ih,
                                                 float* __restrict__ gi) {
    __shared__ _Float16 xs[16][DIN + 8];
    const int row0 = blockIdx.x * 16;
    const int tid  = threadIdx.x;

    for (int i = tid; i < 16 * DIN; i += 768) {
        int m = i >> 8, k = i & 255;
        xs[m][k] = (_Float16)x[(size_t)(row0 + m) * DIN + k];
    }
    __syncthreads();

    const int wave = tid >> 5, lane = tid & 31;
    const int col0 = wave * 16;
    const int m = lane & 15, kk8 = (lane >> 4) * 8;

    v8f acc = {};
    #pragma unroll
    for (int kc = 0; kc < DIN; kc += 32) {
        v16h a;
        #pragma unroll
        for (int v = 0; v < 8; ++v) {
            int k = kc + ((v & 4) << 2) + kk8 + ((v & 3) << 1);
            a[2*v]   = xs[m][k];
            a[2*v+1] = xs[m][k + 1];
        }
        v16h b = load_b16(wih16, DIN, col0, kc, lane);
        acc = __builtin_amdgcn_wmma_f32_16x16x32_f16(false, a, false, b,
                                                     (short)0, acc, false, false);
    }
    const float bias = b_ih[col0 + (lane & 15)];
    const int n = lane & 15, mb = (lane >> 4) * 8;
    #pragma unroll
    for (int j = 0; j < 8; ++j)
        gi[(size_t)(row0 + mb + j) * G3 + col0 + n] = acc[j] + bias;
}

// ---------------------------------------------------------------------------
// Phase 2: sequential GRU scan. ONE 24-wave workgroup; wave g owns gate
// columns [16g,16g+16). w_hh fragments live in registers; h in LDS (f16+f32).
// gi slice for step t+1 is streamed into LDS with global_load_async_to_lds
// during the WMMA phase of step t+1 (single buffer, issue point
// is after the all-readers barrier).
// ---------------------------------------------------------------------------
__global__ __launch_bounds__(768) void gru_scan_kernel(const float* __restrict__ gi,
                                                       const _Float16* __restrict__ whh16,
                                                       const float* __restrict__ b_hh,
                                                       float* __restrict__ hout) {
    __shared__ _Float16 h16[B_][H_ + 8];   // 4.25 KB (padded vs bank conflicts)
    __shared__ float    hf [B_][H_];       // 8 KB
    __shared__ float    gh [B_][G3];       // 24 KB
    __shared__ float    gis[B_][G3];       // 24 KB async-staged gi slice

    const int tid  = threadIdx.x;
    const int wave = tid >> 5, lane = tid & 31;
    const int col0 = wave * 16;

    // Preload w_hh B-fragments (f16, 4 K-chunks) into registers.
    v16h bw[4];
    #pragma unroll
    for (int c = 0; c < 4; ++c)
        bw[c] = load_b16(whh16, H_, col0, c * 32, lane);
    const float bhh = b_hh[col0 + (lane & 15)];

    const unsigned gis_base = (unsigned)(uintptr_t)&gis[0][0]; // LDS address: low 32 bits of the pointer

    // Issue async load of gi slice for t = 0 (2 x b128 per thread = 24 KB).
    {
        #pragma unroll
        for (int kq = 0; kq < 2; ++kq) {
            int c  = tid + kq * 768;           // 16B chunk index in slice
            int b  = c / 96;                   // 96 chunks per row of 384 f32
            int j4 = (c % 96) * 4;
            unsigned goff  = (unsigned)(((b * T_ + 0) * G3 + j4) * 4);
            unsigned laddr = gis_base + (unsigned)((b * G3 + j4) * 4);
            asm volatile("global_load_async_to_lds_b128 %0, %1, %2"
                         :: "v"(laddr), "v"(goff), "s"(gi) : "memory");
        }
    }

    for (int i = tid; i < B_ * H_; i += 768) {
        h16[i >> 7][i & 127] = (_Float16)0.f;
        hf [i >> 7][i & 127] = 0.f;
    }
    __syncthreads();

    const int m = lane & 15, kk8 = (lane >> 4) * 8;
    const int n = lane & 15, mbr = (lane >> 4) * 8;

    for (int t = 0; t < T_; ++t) {
        // gh[:, col0:col0+16] = h @ w_hh^T  (f32 accumulate, 4 WMMA)
        v8f acc = {};
        #pragma unroll
        for (int c = 0; c < 4; ++c) {
            v16h a;
            #pragma unroll
            for (int v = 0; v < 8; ++v) {
                int k = c * 32 + ((v & 4) << 2) + kk8 + ((v & 3) << 1);
                a[2*v]   = h16[m][k];
                a[2*v+1] = h16[m][k + 1];
            }
            acc = __builtin_amdgcn_wmma_f32_16x16x32_f16(false, a, false, bw[c],
                                                         (short)0, acc, false, false);
        }
        #pragma unroll
        for (int j = 0; j < 8; ++j)
            gh[mbr + j][col0 + n] = acc[j] + bhh;

        asm volatile("s_wait_asynccnt 0x0" ::: "memory");  // gi slice t resident
        __syncthreads();

        // Gate combination + state update from LDS only.
        for (int idx = tid; idx < B_ * H_; idx += 768) {
            int b = idx >> 7, j = idx & 127;
            float r  = gis[b][j]           + gh[b][j];
            float z  = gis[b][H_ + j]      + gh[b][H_ + j];
            float gn = gis[b][2 * H_ + j];
            float hn = gh[b][2 * H_ + j];
            r = 1.f / (1.f + __expf(-r));
            z = 1.f / (1.f + __expf(-z));
            float nv    = tanhf(gn + r * hn);
            float hprev = hf[b][j];
            float hnew  = (1.f - z) * nv + z * hprev;
            hf[b][j]  = hnew;
            h16[b][j] = (_Float16)hnew;
            hout[(size_t)(b * T_ + t) * H_ + j] = hnew;
        }
        __syncthreads();

        // All readers done: stream gi slice for t+1 over the next WMMA phase.
        if (t + 1 < T_) {
            #pragma unroll
            for (int kq = 0; kq < 2; ++kq) {
                int c  = tid + kq * 768;
                int b  = c / 96;
                int j4 = (c % 96) * 4;
                unsigned goff  = (unsigned)(((b * T_ + (t + 1)) * G3 + j4) * 4);
                unsigned laddr = gis_base + (unsigned)((b * G3 + j4) * 4);
                asm volatile("global_load_async_to_lds_b128 %0, %1, %2"
                             :: "v"(laddr), "v"(goff), "s"(gi) : "memory");
            }
        }
    }
}

// ---------------------------------------------------------------------------
// Phase 3: y = h @ w_proj^T + b_proj, fused LayerNorm over 128 channels.
// ---------------------------------------------------------------------------
__global__ __launch_bounds__(256) void proj_ln_kernel(const float* __restrict__ hin,
                                                      const _Float16* __restrict__ wp16,
                                                      const float* __restrict__ b_proj,
                                                      const float* __restrict__ gamma,
                                                      const float* __restrict__ beta,
                                                      float* __restrict__ out) {
    __shared__ _Float16 hs[16][H_ + 8];
    __shared__ float    ys[16][H_];
    const int row0 = blockIdx.x * 16;
    const int tid  = threadIdx.x;

    for (int i = tid; i < 16 * H_; i += 256)
        hs[i >> 7][i & 127] = (_Float16)hin[(size_t)(row0 + (i >> 7)) * H_ + (i & 127)];
    __syncthreads();

    const int wave = tid >> 5, lane = tid & 31;
    const int col0 = wave * 16;
    const int m = lane & 15, kk8 = (lane >> 4) * 8;

    v8f acc = {};
    #pragma unroll
    for (int c = 0; c < 4; ++c) {
        v16h a;
        #pragma unroll
        for (int v = 0; v < 8; ++v) {
            int k = c * 32 + ((v & 4) << 2) + kk8 + ((v & 3) << 1);
            a[2*v]   = hs[m][k];
            a[2*v+1] = hs[m][k + 1];
        }
        v16h b = load_b16(wp16, H_, col0, c * 32, lane);
        acc = __builtin_amdgcn_wmma_f32_16x16x32_f16(false, a, false, b,
                                                     (short)0, acc, false, false);
    }
    const float bias = b_proj[col0 + (lane & 15)];
    const int n = lane & 15, mb = (lane >> 4) * 8;
    #pragma unroll
    for (int j = 0; j < 8; ++j)
        ys[mb + j][col0 + n] = acc[j] + bias;
    __syncthreads();

    // LayerNorm: 16 rows x 16 threads; each thread owns 8 columns.
    const int r = tid >> 4, s = tid & 15;
    float sum = 0.f, sq = 0.f;
    #pragma unroll
    for (int c = 0; c < 8; ++c) {
        float v = ys[r][s * 8 + c];
        sum += v; sq += v * v;
    }
    #pragma unroll
    for (int mm = 8; mm >= 1; mm >>= 1) {
        sum += __shfl_xor(sum, mm, 16);
        sq  += __shfl_xor(sq,  mm, 16);
    }
    const float mean = sum * (1.f / 128.f);
    const float var  = sq  * (1.f / 128.f) - mean * mean;
    const float rstd = rsqrtf(var + LN_EPS);
    #pragma unroll
    for (int c = 0; c < 8; ++c) {
        int col = s * 8 + c;
        out[(size_t)(row0 + r) * H_ + col] =
            (ys[r][col] - mean) * rstd * gamma[col] + beta[col];
    }
}

// ---------------------------------------------------------------------------
extern "C" void kernel_launch(void* const* d_in, const int* in_sizes, int n_in,
                              void* d_out, int out_size, void* d_ws, size_t ws_size,
                              hipStream_t stream) {
    const float* x      = (const float*)d_in[0];
    const float* w_ih   = (const float*)d_in[1];
    const float* w_hh   = (const float*)d_in[2];
    const float* b_ih   = (const float*)d_in[3];
    const float* b_hh   = (const float*)d_in[4];
    const float* w_proj = (const float*)d_in[5];
    const float* b_proj = (const float*)d_in[6];
    const float* gamma  = (const float*)d_in[7];
    const float* beta   = (const float*)d_in[8];
    float* out = (float*)d_out;

    char* ws = (char*)d_ws;
    const size_t GI_B = (size_t)B_ * T_ * G3 * sizeof(float);   // 96 MB
    const size_t HB_B = (size_t)B_ * T_ * H_ * sizeof(float);   // 32 MB
    float*    gi    = (float*)ws;
    float*    hbuf  = (float*)(ws + GI_B);
    _Float16* wih16 = (_Float16*)(ws + GI_B + HB_B);
    _Float16* whh16 = wih16 + (size_t)G3 * DIN;
    _Float16* wp16  = whh16 + (size_t)G3 * H_;

    cvt_weights_kernel<<<(G3 * DIN + 255) / 256, 256, 0, stream>>>(
        w_ih, w_hh, w_proj, wih16, whh16, wp16);
    gi_kernel<<<B_ * T_ / 16, 768, 0, stream>>>(x, wih16, b_ih, gi);
    gru_scan_kernel<<<1, 768, 0, stream>>>(gi, whh16, b_hh, hbuf);
    proj_ln_kernel<<<B_ * T_ / 16, 256, 0, stream>>>(hbuf, wp16, b_proj,
                                                     gamma, beta, out);
}
